// GFRB_10591389352537
// MI455X (gfx1250) — matrix operand order repaired and checked by reading, not compiled, not run on hardware
//
#include <hip/hip_runtime.h>
#include <hip/hip_bf16.h>

// ---------------------------------------------------------------------------
// Problem constants (reference): B=4, C=64, H=W=224
// ---------------------------------------------------------------------------
#define BATCH 4
#define CIN   64
#define IMH   224
#define IMW   224
#define HWSZ  (IMH * IMW)             // 50176
#define WTILES (IMW / 32)             // 7 tiles of 32 pixels along W
#define NBLK  (BATCH * IMH * WTILES)  // 6272 blocks per conv pass

typedef __attribute__((ext_vector_type(16))) __bf16 v16bf;
typedef __attribute__((ext_vector_type(8)))  float  v8f;
typedef __attribute__((ext_vector_type(4)))  unsigned int v4u;
typedef __attribute__((ext_vector_type(8)))  int v8i_t;
typedef __attribute__((ext_vector_type(4)))  int v4i_t;

union FragBF { uint4 q[2]; v16bf v; };

// TDM availability (probe-verified builtin; arity differs per toolchain)
#if defined(__has_builtin)
#  if __has_builtin(__builtin_amdgcn_tensor_load_to_lds) && \
      __has_builtin(__builtin_amdgcn_s_wait_tensorcnt)
#    define HAS_TDM 1
#  endif
#endif
#ifndef HAS_TDM
#  define HAS_TDM 0
#endif

// float -> bf16 bits, round-to-nearest-even
__device__ __forceinline__ unsigned short f2bf(float f) {
    union { float f; unsigned u; } v; v.f = f;
    unsigned r = v.u + 0x7FFFu + ((v.u >> 16) & 1u);
    return (unsigned short)(r >> 16);
}

// Generic pointer to __shared__ -> LDS byte address (ISA 10.2: LDS aperture
// addresses use addr[31:0] as the LDS byte offset).
__device__ __forceinline__ unsigned lds_off_of(const void* p) {
    return (unsigned)(unsigned long long)p;
}

// ---------------------------------------------------------------------------
// TDM: async-copy one 32(k) x 64(n) bf16 weight tile (k-step `ks`) from the
// padded bf16 weight matrix wb[64][Ktot] into LDS at lds_byte_off.
// Descriptor packing per CDNA5 ISA.
// ---------------------------------------------------------------------------
__device__ __forceinline__ void tdm_load_b_tile(const unsigned short* wb, int Ktot,
                                                int ks, unsigned lds_byte_off) {
#if HAS_TDM
    unsigned long long ga = (unsigned long long)(const void*)(wb + ks * 32);
    v4u g0;
    g0.x = 1u;                                   // count=1 (valid), user mode
    g0.y = lds_byte_off;                         // lds_addr
    g0.z = (unsigned)(ga & 0xFFFFFFFFu);         // global_addr[31:0]
    g0.w = (unsigned)((ga >> 32) & 0x1FFFFFFu)   // global_addr[56:32]
         | (2u << 30);                           // type = 2 ("image")
    v8i_t g1;
    g1[0] = (1 << 16);                           // data_size=1 (2B); mask=0
    g1[1] = (int)((unsigned)(Ktot & 0xFFFF) << 16);  // tensor_dim0[15:0]
    g1[2] = (int)(((unsigned)Ktot >> 16) | (64u << 16)); // dim0 hi | tensor_dim1=64
    g1[3] = (int)(32u << 16);                    // tile_dim0 = 32
    g1[4] = 64;                                  // tile_dim1 = 64, tile_dim2 = 0
    g1[5] = Ktot;                                // tensor_dim0_stride[31:0]
    g1[6] = 0;                                   // stride hi | dim1_stride lo
    g1[7] = 0;
    v4i_t z4 = {0, 0, 0, 0};
#if __clang_major__ >= 23
    v8i_t z8 = {0, 0, 0, 0, 0, 0, 0, 0};
    __builtin_amdgcn_tensor_load_to_lds(g0, g1, z4, z4, z8, 0);
#else
    __builtin_amdgcn_tensor_load_to_lds(g0, g1, z4, z4, 0);
#endif
#endif
}

// ---------------------------------------------------------------------------
// Fused elementwise prologue over x:
//   out  = x*(dw1_w+1) + dw1_b            (identity + dw branch, fp32)
//   im2  = bf16(x*dw2_w + dw2_b)          (input image for offset conv)
//   im3  = bf16(x*dw3_w + dw3_b)          (input image for asym1 conv)
// ---------------------------------------------------------------------------
__global__ __launch_bounds__(256)
void prep_ew_kernel(const float* __restrict__ x,
                    const float* __restrict__ s1, const float* __restrict__ t1,
                    const float* __restrict__ s2, const float* __restrict__ t2,
                    const float* __restrict__ s3, const float* __restrict__ t3,
                    float* __restrict__ out,
                    unsigned short* __restrict__ im2,
                    unsigned short* __restrict__ im3) {
    int i = (blockIdx.x * 256 + threadIdx.x) * 4;   // exact cover
    int c = (i / HWSZ) & 63;
    float a1 = s1[c] + 1.0f, b1 = t1[c];
    float a2 = s2[c],        b2 = t2[c];
    float a3 = s3[c],        b3 = t3[c];
    float4 v = *(const float4*)(x + i);
    float4 o;
    o.x = v.x * a1 + b1; o.y = v.y * a1 + b1;
    o.z = v.z * a1 + b1; o.w = v.w * a1 + b1;
    *(float4*)(out + i) = o;
    uint2 p2, p3;
    p2.x = (unsigned)f2bf(v.x * a2 + b2) | ((unsigned)f2bf(v.y * a2 + b2) << 16);
    p2.y = (unsigned)f2bf(v.z * a2 + b2) | ((unsigned)f2bf(v.w * a2 + b2) << 16);
    p3.x = (unsigned)f2bf(v.x * a3 + b3) | ((unsigned)f2bf(v.y * a3 + b3) << 16);
    p3.y = (unsigned)f2bf(v.z * a3 + b3) | ((unsigned)f2bf(v.w * a3 + b3) << 16);
    *(uint2*)(im2 + i) = p2;
    *(uint2*)(im3 + i) = p3;
}

// ---------------------------------------------------------------------------
// Weight prep: src (Cout, Ktot) fp32 -> dst (64, Ktot) bf16, zero-padded rows.
// ---------------------------------------------------------------------------
__global__ __launch_bounds__(256)
void prep_w_kernel(const float* __restrict__ src, int Cout, int Ktot,
                   unsigned short* __restrict__ dst) {
    int idx = blockIdx.x * 256 + threadIdx.x;  // grid sized to 64*Ktot exactly
    int n = idx / Ktot;
    dst[idx] = (n < Cout) ? f2bf(src[idx]) : (unsigned short)0;
}

// ---------------------------------------------------------------------------
// Im2col conv as WMMA GEMM (bf16 in, f32 acc).
//   inb : (B,64,H,W) bf16 image (dw 1x1 already fused in)
//   wb  : (64, Ktot) bf16, zero-padded; staged per k-step via TDM double buffer
// Block: 256 thr = 8 waves = (2 M x 4 N) = 32 pixels x 64 channels.
// mode: 0 = store f32 (+bias), 1 = accumulate into f32, 2 = store bf16.
// ---------------------------------------------------------------------------
template<int KH, int KW, int PH, int PW>
__global__ __launch_bounds__(256)
void conv_wmma_kernel(const unsigned short* __restrict__ inb,
                      const unsigned short* __restrict__ wb,
                      const float* __restrict__ bias, int Cout,
                      float* __restrict__ outf,
                      unsigned short* __restrict__ outb, int mode) {
    constexpr int KPC    = KH * KW;
    constexpr int KTOT   = CIN * KPC;
    constexpr int KSTEPS = KTOT / 32;

    __shared__ unsigned short aT[32][32];      // [pix][kk]
    __shared__ unsigned short bT[2][64][32];   // double-buffered [n][kk]

    const int tid  = threadIdx.x;
    const int lane = tid & 31;
    const int wv   = tid >> 5;
    const int m0   = (wv & 1) * 16;
    const int n0   = (wv >> 1) * 16;

    const int tw     = blockIdx.x % WTILES;
    const int h      = (blockIdx.x / WTILES) % IMH;
    const int b      = blockIdx.x / (WTILES * IMH);
    const int w_base = tw * 32;

    const int pix = tid & 31;    // A staging: lane -> pixel (coalesced u16)
    const int kk0 = tid >> 5;

#if HAS_TDM
    if (wv == 0) tdm_load_b_tile(wb, KTOT, 0, lds_off_of(&bT[0][0][0]));
#endif

    v8f acc = {};
    #pragma unroll 2
    for (int ks = 0; ks < KSTEPS; ++ks) {
        __syncthreads();   // prior reads of aT / bT[(ks+1)&1] complete
#if HAS_TDM
        if (wv == 0 && ks + 1 < KSTEPS)
            tdm_load_b_tile(wb, KTOT, ks + 1, lds_off_of(&bT[(ks + 1) & 1][0][0]));
#else
        #pragma unroll
        for (int j = 0; j < 8; ++j) {
            int kk = tid & 31;
            int n  = (tid >> 5) + j * 8;
            bT[ks & 1][n][kk] = wb[n * KTOT + ks * 32 + kk];
        }
#endif
        // ---- stage A: 32 pix x 32 kk, bf16 image loads ----
        #pragma unroll
        for (int j = 0; j < 4; ++j) {
            int kk = kk0 + j * 8;
            int kg = ks * 32 + kk;
            int c  = kg / KPC;            // compile-time divisor
            int t  = kg - c * KPC;
            int ky = t / KW;
            int kx = t - ky * KW;
            int y  = h + ky - PH;
            int x  = w_base + pix + kx - PW;
            unsigned short v = 0;
            if ((unsigned)y < IMH && (unsigned)x < IMW)
                v = inb[((b * CIN + c) * IMH + y) * IMW + x];
            aT[pix][kk] = v;
        }
#if HAS_TDM
        if (ks + 1 < KSTEPS) { if (wv == 0) __builtin_amdgcn_s_wait_tensorcnt(1); }
        else                 { if (wv == 0) __builtin_amdgcn_s_wait_tensorcnt(0); }
#endif
        __syncthreads();   // A staged + B(ks) visible to all waves

        // ---- fragment loads per ISA VGPR layouts, then WMMA ----
        FragBF af, bf;
        {
            int row = m0 + (lane & 15);
            int klo = (lane < 16) ? 0 : 8;       // A: K 0-7/16-23 vs 8-15/24-31
            af.q[0] = *(const uint4*)&aT[row][klo];
            af.q[1] = *(const uint4*)&aT[row][klo + 16];
        }
        {
            int n     = n0 + (lane & 15);
            int kbase = (lane < 16) ? 0 : 8;     // B: K 0-7/16-23 vs 8-15/24-31
            bf.q[0] = *(const uint4*)&bT[ks & 1][n][kbase];
            bf.q[1] = *(const uint4*)&bT[ks & 1][n][kbase + 16];
        }
        acc = __builtin_amdgcn_wmma_f32_16x16x32_bf16(
                false, af.v, false, bf.v, (short)0, acc, false, false);
    }

    // ---- epilogue: D layout  lane<16: M=v, N=lane ; lane>=16: M=v+8 ----
    int ch = n0 + (lane & 15);
    if (ch < Cout) {
        float bs = bias[ch];
        #pragma unroll
        for (int v = 0; v < 8; ++v) {
            int pl = m0 + v + ((lane < 16) ? 0 : 8);
            int wp = w_base + pl;
            size_t o = ((size_t)(b * Cout + ch) * IMH + h) * IMW + wp;
            float r = acc[v] + bs;
            if (mode == 0)      outf[o] = r;
            else if (mode == 1) outf[o] += r;
            else                outb[o] = f2bf(r);
        }
    }
}

// ---------------------------------------------------------------------------
// Deformable conv: bilinear-sample A[32 pix][576] (k = c*9 + tap) into LDS,
// then 18 bf16 WMMA k-steps vs TDM-staged def_w tiles. Accumulates into out.
// ---------------------------------------------------------------------------
__global__ __launch_bounds__(256)
void deform_wmma_kernel(const float* __restrict__ x,
                        const float* __restrict__ off,          // (B,18,H,W) f32
                        const unsigned short* __restrict__ wb,  // (64,576) bf16
                        const float* __restrict__ bias,
                        float* __restrict__ out) {
    __shared__ unsigned short S[32][576];     // sampled im2col (36 KB)
    __shared__ unsigned short bT[2][64][32];  // double-buffered weight tiles

    const int tid  = threadIdx.x;
    const int lane = tid & 31;
    const int wv   = tid >> 5;
    const int m0   = (wv & 1) * 16;
    const int n0   = (wv >> 1) * 16;

    const int tw     = blockIdx.x % WTILES;
    const int h      = (blockIdx.x / WTILES) % IMH;
    const int b      = blockIdx.x / (WTILES * IMH);
    const int w_base = tw * 32;

#if HAS_TDM
    if (wv == 0) tdm_load_b_tile(wb, 576, 0, lds_off_of(&bT[0][0][0]));
#endif

    // ---- sampling stage: 32 pix x 9 taps = 288 items over 256 threads ----
    for (int item = tid; item < 288; item += 256) {
        int pix = item / 9;
        int tap = item - pix * 9;
        int wp  = w_base + pix;
        int ky  = tap / 3 - 1;
        int kx  = tap % 3 - 1;
        float dy = off[((b * 18 + 2 * tap    ) * IMH + h) * IMW + wp];
        float dx = off[((b * 18 + 2 * tap + 1) * IMH + h) * IMW + wp];
        float py = (float)(h  + ky) + dy;
        float px = (float)(wp + kx) + dx;
        float y0f = floorf(py), x0f = floorf(px);
        float fy = py - y0f,    fx = px - x0f;
        int y0 = (int)y0f, x0 = (int)x0f;
        int y1 = y0 + 1,   x1 = x0 + 1;
        bool vy0 = (y0 >= 0) & (y0 < IMH), vy1 = (y1 >= 0) & (y1 < IMH);
        bool vx0 = (x0 >= 0) & (x0 < IMW), vx1 = (x1 >= 0) & (x1 < IMW);
        float w00 = (vy0 & vx0) ? (1.f - fy) * (1.f - fx) : 0.f;
        float w01 = (vy0 & vx1) ? (1.f - fy) * fx         : 0.f;
        float w10 = (vy1 & vx0) ? fy * (1.f - fx)         : 0.f;
        float w11 = (vy1 & vx1) ? fy * fx                 : 0.f;
        int y0c = min(max(y0, 0), IMH - 1), y1c = min(max(y1, 0), IMH - 1);
        int x0c = min(max(x0, 0), IMW - 1), x1c = min(max(x1, 0), IMW - 1);
        int i00 = y0c * IMW + x0c, i01 = y0c * IMW + x1c;
        int i10 = y1c * IMW + x0c, i11 = y1c * IMW + x1c;
        const float* xb = x + (size_t)b * CIN * HWSZ;
        for (int c = 0; c < CIN; ++c) {
            const float* xc = xb + (size_t)c * HWSZ;
            float s = w00 * xc[i00] + w01 * xc[i01]
                    + w10 * xc[i10] + w11 * xc[i11];
            S[pix][c * 9 + tap] = f2bf(s);
        }
    }

    v8f acc = {};
    #pragma unroll 2
    for (int ks = 0; ks < 18; ++ks) {
        __syncthreads();
#if HAS_TDM
        if (wv == 0 && ks + 1 < 18)
            tdm_load_b_tile(wb, 576, ks + 1, lds_off_of(&bT[(ks + 1) & 1][0][0]));
        if (ks + 1 < 18) { if (wv == 0) __builtin_amdgcn_s_wait_tensorcnt(1); }
        else             { if (wv == 0) __builtin_amdgcn_s_wait_tensorcnt(0); }
#else
        #pragma unroll
        for (int j = 0; j < 8; ++j) {
            int kk = tid & 31;
            int n  = (tid >> 5) + j * 8;
            bT[ks & 1][n][kk] = wb[n * 576 + ks * 32 + kk];
        }
#endif
        __syncthreads();

        FragBF af, bf;
        {
            int row = m0 + (lane & 15);
            int klo = ks * 32 + ((lane < 16) ? 0 : 8);
            af.q[0] = *(const uint4*)&S[row][klo];
            af.q[1] = *(const uint4*)&S[row][klo + 16];
        }
        {
            int n     = n0 + (lane & 15);
            int kbase = (lane < 16) ? 0 : 8;
            bf.q[0] = *(const uint4*)&bT[ks & 1][n][kbase];
            bf.q[1] = *(const uint4*)&bT[ks & 1][n][kbase + 16];
        }
        acc = __builtin_amdgcn_wmma_f32_16x16x32_bf16(
                false, af.v, false, bf.v, (short)0, acc, false, false);
    }

    int ch = n0 + (lane & 15);
    float bs = bias[ch];
    #pragma unroll
    for (int v = 0; v < 8; ++v) {
        int pl = m0 + v + ((lane < 16) ? 0 : 8);
        int wp = w_base + pl;
        size_t o = ((size_t)(b * CIN + ch) * IMH + h) * IMW + wp;
        out[o] += acc[v] + bs;
    }
}

// ---------------------------------------------------------------------------
// Launch
// ---------------------------------------------------------------------------
extern "C" void kernel_launch(void* const* d_in, const int* in_sizes, int n_in,
                              void* d_out, int out_size, void* d_ws, size_t ws_size,
                              hipStream_t stream) {
    const float* x       = (const float*)d_in[0];
    const float* dw1_w   = (const float*)d_in[1];
    const float* dw1_b   = (const float*)d_in[2];
    const float* dw2_w   = (const float*)d_in[3];
    const float* dw2_b   = (const float*)d_in[4];
    const float* dw3_w   = (const float*)d_in[5];
    const float* dw3_b   = (const float*)d_in[6];
    const float* asym1_w = (const float*)d_in[7];
    const float* asym1_b = (const float*)d_in[8];
    const float* asym2_w = (const float*)d_in[9];
    const float* asym2_b = (const float*)d_in[10];
    const float* off_w   = (const float*)d_in[11];
    const float* off_b   = (const float*)d_in[12];
    const float* def_w   = (const float*)d_in[13];
    const float* def_b   = (const float*)d_in[14];
    float* out = (float*)d_out;

    // ---- workspace carve-up (all chunks multiple of 256 B) ----
    char* w = (char*)d_ws;
    const size_t sz_off  = (size_t)BATCH * 18 * HWSZ * 4;   // offsets, f32
    const size_t sz_img  = (size_t)BATCH * CIN * HWSZ * 2;  // bf16 image
    float*          ws_off = (float*)w;                      w += sz_off;
    unsigned short* im2    = (unsigned short*)w;             w += sz_img;
    unsigned short* im3    = (unsigned short*)w;             w += sz_img;
    unsigned short* a1b    = (unsigned short*)w;             w += sz_img;
    unsigned short* wb_off = (unsigned short*)w;             w += (size_t)64 * 576 * 2;
    unsigned short* wb_def = (unsigned short*)w;             w += (size_t)64 * 576 * 2;
    unsigned short* wb_a1  = (unsigned short*)w;             w += (size_t)64 * 320 * 2;
    unsigned short* wb_a2  = (unsigned short*)w;             w += (size_t)64 * 320 * 2;

    const int n = BATCH * CIN * HWSZ;

    // 1) fused elementwise: residual+dw1 into out; dw2(x), dw3(x) as bf16 images
    prep_ew_kernel<<<n / (256 * 4), 256, 0, stream>>>(
        x, dw1_w, dw1_b, dw2_w, dw2_b, dw3_w, dw3_b, out, im2, im3);

    // 2) weight conversion to padded bf16 (TDM source matrices)
    prep_w_kernel<<<(64 * 576) / 256, 256, 0, stream>>>(off_w,   18, 576, wb_off);
    prep_w_kernel<<<(64 * 576) / 256, 256, 0, stream>>>(def_w,   64, 576, wb_def);
    prep_w_kernel<<<(64 * 320) / 256, 256, 0, stream>>>(asym1_w, 64, 320, wb_a1);
    prep_w_kernel<<<(64 * 320) / 256, 256, 0, stream>>>(asym2_w, 64, 320, wb_a2);

    // 3) offset conv: 3x3 pad 1 on dw2 image -> ws_off (f32)
    conv_wmma_kernel<3, 3, 1, 1><<<NBLK, 256, 0, stream>>>(
        im2, wb_off, off_b, 18, ws_off, nullptr, 0);

    // 4) deformable conv, accumulate into out
    deform_wmma_kernel<<<NBLK, 256, 0, stream>>>(x, ws_off, wb_def, def_b, out);

    // 5) asym 5x1 conv (pad 2,0) on dw3 image -> a1 bf16 image
    conv_wmma_kernel<5, 1, 2, 0><<<NBLK, 256, 0, stream>>>(
        im3, wb_a1, asym1_b, 64, nullptr, a1b, 2);

    // 6) asym 1x5 conv (pad 0,2) on a1, accumulate into out
    conv_wmma_kernel<1, 5, 0, 2><<<NBLK, 256, 0, stream>>>(
        a1b, wb_a2, asym2_b, 64, out, nullptr, 1);
}
